// Decoder_att_37890201485562
// MI455X (gfx1250) — matrix operand order repaired and checked by reading, not compiled, not run on hardware
//
#include <hip/hip_runtime.h>
#include <hip/hip_bf16.h>
#include <cstdint>
#include <cstddef>

// ---------------------------------------------------------------------------
// Tacotron2-style decoder with triple location-sensitive attention, MI455X.
// Strategy: bf16 WMMA (v_wmma_f32_16x16x32_bf16) for the two per-step LSTM
// gate GEMMs (weights packed to bf16 once; ~33MB -> resident in 192MB L2),
// VALU kernels for attention / elementwise, 6 launches per recurrent step.
// Round 3: KTOT is a compile-time template parameter (1536 / 2560), K-loop is
// a branch-free peeled ping-pong pipeline (no guards, no register-copy
// fallback blocks), LDS sized per instantiation.
// ---------------------------------------------------------------------------

typedef __attribute__((ext_vector_type(16))) __bf16 v16bf;
typedef __attribute__((ext_vector_type(8)))  __bf16 v8bf;
typedef __attribute__((ext_vector_type(8)))  float  v8f;

#define BB    8
#define TENC  256
#define TDEC  128
#define NMELC 80
#define E1C   64
#define E2C   128
#define E3C   64
#define PPC   256
#define AAC   128
#define FFC   32
#define KKC   31
#define HC    1024      // HATT == HDEC
#define GATEN 4096      // 4*H
#define SPKC  256

// ---------------------------------------------------------------------------
// one-time weight pack: [W_ih | W_hh] rows -> bf16, row-major [n][k]
// ---------------------------------------------------------------------------
__global__ void pack2_kernel(const float* __restrict__ A, int la,
                             const float* __restrict__ Bm, int lb,
                             __bf16* __restrict__ dst, int nrows) {
  const int Kt = la + lb;
  size_t idx = (size_t)blockIdx.x * 256 + threadIdx.x;
  if (idx >= (size_t)nrows * Kt) return;
  int n = (int)(idx / Kt);
  int k = (int)(idx % Kt);
  float v = (k < la) ? A[(size_t)n * la + k] : Bm[(size_t)n * lb + (k - la)];
  dst[idx] = (__bf16)v;
}

__global__ void zero_kernel(float* p, int n) {
  int i = blockIdx.x * 256 + threadIdx.x;
  if (i < n) p[i] = 0.f;
}

// ---------------------------------------------------------------------------
// prenet (all 128 steps at once): pre = relu(relu(tf_in @ w1) @ w2)
// tf_in[t][b][k] = (t==0) ? 0 : decoder_inputs[b][k][t-1]
// ---------------------------------------------------------------------------
__global__ void prenet1_kernel(const float* __restrict__ din,
                               const float* __restrict__ w1,
                               float* __restrict__ h1) {
  int idx = blockIdx.x * 256 + threadIdx.x;
  if (idx >= TDEC * BB * PPC) return;
  int t = idx / (BB * PPC);
  int r = idx % (BB * PPC);
  int b = r / PPC, j = r % PPC;
  float acc = 0.f;
  if (t > 0) {
    for (int k = 0; k < NMELC; ++k)
      acc += din[((size_t)b * NMELC + k) * TDEC + (t - 1)] * w1[k * PPC + j];
  }
  h1[idx] = fmaxf(acc, 0.f);
}

__global__ void prenet2_kernel(const float* __restrict__ h1,
                               const float* __restrict__ w2,
                               float* __restrict__ pre) {
  int idx = blockIdx.x * 256 + threadIdx.x;
  if (idx >= TDEC * BB * PPC) return;
  int t = idx / (BB * PPC);
  int r = idx % (BB * PPC);
  int b = r / PPC, j = r % PPC;
  const float* hrow = h1 + (size_t)(t * BB + b) * PPC;
  float acc = 0.f;
  for (int k = 0; k < PPC; ++k) acc += hrow[k] * w2[k * PPC + j];
  pre[idx] = fmaxf(acc, 0.f);
}

// pmem_i = memory_i @ Wm_i   -> (B, TENC, A)
__global__ void pmem_kernel(const float* __restrict__ mem,
                            const float* __restrict__ Wm,
                            float* __restrict__ pm, int E) {
  int idx = blockIdx.x * 256 + threadIdx.x;
  if (idx >= BB * TENC * AAC) return;
  int bt = idx / AAC, a = idx % AAC;
  float acc = 0.f;
  for (int e = 0; e < E; ++e) acc += mem[(size_t)bt * E + e] * Wm[e * AAC + a];
  pm[idx] = acc;
}

// ---------------------------------------------------------------------------
// WMMA LSTM gates: gates(8x4096) = concat(segs)(8xKTOT) @ Wcat^T (bf16,
// f32 acc). A staged in LDS [m][k] bf16 with a zero-pad row for lanes m>=8
// (branch-free loads); one wave per 16-wide N tile; branch-free peeled
// ping-pong K-loop of KTOT/32 steps (trip count is compile-time).
// ---------------------------------------------------------------------------
struct Segs { const float* p[6]; int len[6]; };

template <int KTOT>
__global__ __launch_bounds__(256) void wmma_gates_kernel(
    const __bf16* __restrict__ Wcat, Segs segs,
    const float* __restrict__ bih, const float* __restrict__ bhh,
    float* __restrict__ gates) {
  constexpr int ZP = BB * KTOT;            // zero-pad row offset
  __shared__ __bf16 sA[BB * KTOT + 32];
  const int tid = threadIdx.x;

  // stage the concat activation vector (8 rows) into LDS as bf16
  for (int idx = tid; idx < BB * KTOT; idx += 256) {
    int b = idx / KTOT;
    int k = idx - b * KTOT;
    float val = 0.f;
    int kr = k;
#pragma unroll
    for (int s = 0; s < 6; ++s) {
      int L = segs.len[s];
      if (kr >= 0 && kr < L) val = segs.p[s][b * L + kr];
      kr -= L;
    }
    sA[b * KTOT + k] = (__bf16)val;
  }
  if (tid < 32) sA[ZP + tid] = (__bf16)0.0f;   // zero pad row
  __syncthreads();

  const int lane  = tid & 31;
  const int wave  = tid >> 5;
  const int half  = lane >> 4;    // K-half selector: k = 16*half + s in both A and B
  const int m     = lane & 15;    // A row (batch) / B column-in-tile
  const int ntile = blockIdx.x * 8 + wave;
  const __bf16* wrow = Wcat + (size_t)(ntile * 16 + m) * KTOT;

  // branch-free A addressing: lanes m>=8 read the zero pad with stride 0
  const int rbase = (m < 8) ? m * KTOT : ZP;
  const int rstep = (m < 8) ? 32 : 0;

  v8f c;
#pragma unroll
  for (int i = 0; i < 8; ++i) c[i] = 0.f;

  constexpr int ksteps = KTOT >> 5;        // 48 or 80: even, >= 2
  static_assert((ksteps & 1) == 0 && ksteps >= 2, "ksteps must be even");

  auto loadA = [&](int kk) -> v16bf {
    return *(const v16bf*)(sA + rbase + kk * rstep + half * 16);
  };
  auto loadB = [&](int kk) -> v16bf {
    return *(const v16bf*)(wrow + kk * 32 + half * 16);
  };

  // peeled ping-pong pipeline: the loop body is branch-free and always keeps
  // one (A,B) tile pair in flight ahead of the WMMA that consumes it.
  v16bf a0 = loadA(0), b0 = loadB(0);
  v16bf a1, b1;
  for (int kk = 0; kk + 2 < ksteps; kk += 2) {
    a1 = loadA(kk + 1);
    b1 = loadB(kk + 1);
    __builtin_prefetch(wrow + (kk + 4) * 32 + half * 16, 0, 3);
    c = __builtin_amdgcn_wmma_f32_16x16x32_bf16(false, a0, false, b0,
                                                (short)0, c, false, false);
    a0 = loadA(kk + 2);
    b0 = loadB(kk + 2);
    c = __builtin_amdgcn_wmma_f32_16x16x32_bf16(false, a1, false, b1,
                                                (short)0, c, false, false);
  }
  a1 = loadA(ksteps - 1);
  b1 = loadB(ksteps - 1);
  c = __builtin_amdgcn_wmma_f32_16x16x32_bf16(false, a0, false, b0,
                                              (short)0, c, false, false);
  c = __builtin_amdgcn_wmma_f32_16x16x32_bf16(false, a1, false, b1,
                                              (short)0, c, false, false);

  // D: lanes 0-15, VGPR r -> M=r (batch rows 0..7); lanes 16-31 are padding
  if (half == 0) {
    int n = ntile * 16 + m;
    float bias = bih[n] + bhh[n];
#pragma unroll
    for (int r = 0; r < 8; ++r) gates[(size_t)r * GATEN + n] = c[r] + bias;
  }
}

// elementwise LSTM cell update (gate order i,f,g,o)
__global__ void lstm_update_kernel(const float* __restrict__ gates,
                                   float* __restrict__ h, float* __restrict__ c) {
  int idx = blockIdx.x * 256 + threadIdx.x;
  if (idx >= BB * HC) return;
  int b = idx / HC, j = idx % HC;
  const float* g = gates + (size_t)b * GATEN;
  float gi = g[j], gf = g[HC + j], gg = g[2 * HC + j], go = g[3 * HC + j];
  float si = 1.f / (1.f + __expf(-gi));
  float sf = 1.f / (1.f + __expf(-gf));
  float so = 1.f / (1.f + __expf(-go));
  float c2 = sf * c[idx] + si * tanhf(gg);
  c[idx] = c2;
  h[idx] = so * tanhf(c2);
}

// ---------------------------------------------------------------------------
// location-sensitive attention: one block per (attention, batch)
// ---------------------------------------------------------------------------
struct AttnP {
  const float* Wq; const float* vv; const float* lconv; const float* ldense;
  const float* mem; const float* pmem;
  float* wprev; float* wcum; float* x; float* aout;
  int E; int pad;
};
struct Attn3 { AttnP p[3]; };

__global__ __launch_bounds__(256) void attention_kernel(Attn3 prm,
                                                        const float* __restrict__ ha,
                                                        int tstep) {
  const int b  = blockIdx.x & 7;
  const int ai = blockIdx.x >> 3;
  const AttnP P = prm.p[ai];
  const int tid = threadIdx.x;

  __shared__ float loc[FFC * TENC];
  __shared__ float wp[TENC], wc[TENC], qv[AAC], ev[TENC], wnew[TENC], red[TENC];

  wp[tid] = P.wprev[b * TENC + tid];
  wc[tid] = P.wcum [b * TENC + tid];
  if (tid < AAC) {  // query projection h @ Wq
    float acc = 0.f;
    for (int h = 0; h < HC; ++h) acc += ha[b * HC + h] * P.Wq[h * AAC + tid];
    qv[tid] = acc;
  }
  __syncthreads();

  // location conv over [w_prev, w_cum] (correlation, SAME padding)
  for (int o = tid; o < FFC * TENC; o += 256) {
    int f = o >> 8, t = o & 255;
    const float* k0 = P.lconv + f * (2 * KKC);
    const float* k1 = k0 + KKC;
    float acc = 0.f;
#pragma unroll
    for (int k = 0; k < KKC; ++k) {
      int tt = t + k - (KKC - 1) / 2;
      if (tt >= 0 && tt < TENC) acc += wp[tt] * k0[k] + wc[tt] * k1[k];
    }
    loc[o] = acc;
  }
  __syncthreads();

  // energies: e[t] = sum_a tanh(q[a] + (loc^T @ ldense)[t][a] + pmem) * v[a]
  {
    int t = tid;
    const float* pm = P.pmem + (size_t)(b * TENC + t) * AAC;
    float acc = 0.f;
    for (int a = 0; a < AAC; ++a) {
      float pl = 0.f;
#pragma unroll
      for (int f = 0; f < FFC; ++f) pl += loc[f * TENC + t] * P.ldense[f * AAC + a];
      acc += tanhf(qv[a] + pl + pm[a]) * P.vv[a];
    }
    ev[t] = acc;
  }
  __syncthreads();

  // softmax over T (masks are all-false in this problem)
  red[tid] = ev[tid];
  __syncthreads();
  for (int s = 128; s > 0; s >>= 1) {
    if (tid < s) red[tid] = fmaxf(red[tid], red[tid + s]);
    __syncthreads();
  }
  float mx = red[0];
  __syncthreads();
  float ex = __expf(ev[tid] - mx);
  red[tid] = ex;
  __syncthreads();
  for (int s = 128; s > 0; s >>= 1) {
    if (tid < s) red[tid] += red[tid + s];
    __syncthreads();
  }
  float inv = 1.f / red[0];
  __syncthreads();
  wnew[tid] = ex * inv;
  __syncthreads();

  // context = w @ mem
  if (tid < P.E) {
    const float* mb = P.mem + (size_t)b * TENC * P.E + tid;
    float acc = 0.f;
    for (int t = 0; t < TENC; ++t) acc += wnew[t] * mb[(size_t)t * P.E];
    P.x[b * P.E + tid] = acc;
  }
  // state + output weights
  P.wprev[b * TENC + tid] = wnew[tid];
  P.wcum [b * TENC + tid] = wc[tid] + wnew[tid];
  P.aout[((size_t)b * TDEC + tstep) * TENC + tid] = wnew[tid];
}

// ---------------------------------------------------------------------------
// output projections: mel (1536->80) and stop gate (1280->1)
// ---------------------------------------------------------------------------
__global__ void proj_kernel(const float* __restrict__ hd, const float* __restrict__ x1,
                            const float* __restrict__ x2, const float* __restrict__ x3,
                            const float* __restrict__ spk,
                            const float* __restrict__ Wp, const float* __restrict__ bp,
                            const float* __restrict__ Wg, const float* __restrict__ bg,
                            float* __restrict__ outmel, float* __restrict__ outgate,
                            int tstep) {
  const int b = blockIdx.x;
  const int n = threadIdx.x;
  if (n < NMELC) {
    float acc = bp[n];
    int k = 0;
    for (int j = 0; j < HC;   ++j, ++k) acc += hd [b * HC   + j] * Wp[k * NMELC + n];
    for (int j = 0; j < E1C;  ++j, ++k) acc += x1 [b * E1C  + j] * Wp[k * NMELC + n];
    for (int j = 0; j < E2C;  ++j, ++k) acc += x2 [b * E2C  + j] * Wp[k * NMELC + n];
    for (int j = 0; j < E3C;  ++j, ++k) acc += x3 [b * E3C  + j] * Wp[k * NMELC + n];
    for (int j = 0; j < SPKC; ++j, ++k) acc += spk[b * SPKC + j] * Wp[k * NMELC + n];
    outmel[((size_t)b * NMELC + n) * TDEC + tstep] = acc;
  } else if (n == NMELC) {
    float acc = bg[0];
    int k = 0;
    for (int j = 0; j < HC;  ++j, ++k) acc += hd[b * HC  + j] * Wg[k];
    for (int j = 0; j < E1C; ++j, ++k) acc += x1[b * E1C + j] * Wg[k];
    for (int j = 0; j < E2C; ++j, ++k) acc += x2[b * E2C + j] * Wg[k];
    for (int j = 0; j < E3C; ++j, ++k) acc += x3[b * E3C + j] * Wg[k];
    outgate[b * TDEC + tstep] = acc;
  }
}

// ---------------------------------------------------------------------------
// host driver
// ---------------------------------------------------------------------------
extern "C" void kernel_launch(void* const* d_in, const int* in_sizes, int n_in,
                              void* d_out, int out_size, void* d_ws, size_t ws_size,
                              hipStream_t stream) {
  (void)in_sizes; (void)n_in; (void)out_size; (void)ws_size;

  const float* dec_inputs = (const float*)d_in[0];
  const float* mem1 = (const float*)d_in[1];
  const float* mem2 = (const float*)d_in[2];
  const float* mem3 = (const float*)d_in[3];
  const float* spk  = (const float*)d_in[4];
  // d_in[5..7] masks: all-false, ignored
  const float* pw1    = (const float*)d_in[8];
  const float* pw2    = (const float*)d_in[9];
  const float* W_ih_a = (const float*)d_in[10];
  const float* W_hh_a = (const float*)d_in[11];
  const float* b_ih_a = (const float*)d_in[12];
  const float* b_hh_a = (const float*)d_in[13];
  const float* Wq[3] = {(const float*)d_in[14], (const float*)d_in[19], (const float*)d_in[24]};
  const float* Wm[3] = {(const float*)d_in[15], (const float*)d_in[20], (const float*)d_in[25]};
  const float* vv[3] = {(const float*)d_in[16], (const float*)d_in[21], (const float*)d_in[26]};
  const float* lc[3] = {(const float*)d_in[17], (const float*)d_in[22], (const float*)d_in[27]};
  const float* ld[3] = {(const float*)d_in[18], (const float*)d_in[23], (const float*)d_in[28]};
  const float* W_ih_d = (const float*)d_in[29];
  const float* W_hh_d = (const float*)d_in[30];
  const float* b_ih_d = (const float*)d_in[31];
  const float* b_hh_d = (const float*)d_in[32];
  const float* Wp = (const float*)d_in[33];
  const float* bp = (const float*)d_in[34];
  const float* Wg = (const float*)d_in[35];
  const float* bg = (const float*)d_in[36];

  // workspace carve-up (all sizes are multiples of 256B -> exact packing)
  char* ws = (char*)d_ws;
  size_t off = 0;
  auto take = [&](size_t bytes) -> char* {
    char* r = ws + off;
    off = (off + bytes + 255) & ~(size_t)255;
    return r;
  };
  const int KA = 1536, KD = 2560;
  __bf16* WcatA = (__bf16*)take((size_t)GATEN * KA * sizeof(__bf16));
  __bf16* WcatD = (__bf16*)take((size_t)GATEN * KD * sizeof(__bf16));
  float* h1  = (float*)take((size_t)TDEC * BB * PPC * 4);
  float* pre = (float*)take((size_t)TDEC * BB * PPC * 4);
  float* pm[3];
  pm[0] = (float*)take((size_t)BB * TENC * AAC * 4);
  pm[1] = (float*)take((size_t)BB * TENC * AAC * 4);
  pm[2] = (float*)take((size_t)BB * TENC * AAC * 4);
  float* ha = (float*)take(BB * HC * 4);
  float* ca = (float*)take(BB * HC * 4);
  float* hd = (float*)take(BB * HC * 4);
  float* cd = (float*)take(BB * HC * 4);
  float* x1 = (float*)take(BB * E1C * 4);
  float* x2 = (float*)take(BB * E2C * 4);
  float* x3 = (float*)take(BB * E3C * 4);
  float* wprev[3]; float* wcum[3];
  for (int i = 0; i < 3; ++i) {
    wprev[i] = (float*)take(BB * TENC * 4);
    wcum[i]  = (float*)take(BB * TENC * 4);
  }
  float* gates = (float*)take((size_t)BB * GATEN * 4);
  // contiguous per-step state block starting at ha
  const int zcount = 4 * BB * HC + BB * (E1C + E2C + E3C) + 6 * BB * TENC + BB * GATEN;

  // output regions (mels, gates, a1, a2, a3)
  float* out = (float*)d_out;
  float* outmel  = out;
  float* outgate = out + BB * NMELC * TDEC;                 // 81920
  float* outa[3];
  outa[0] = outgate + BB * TDEC;                            // 82944
  outa[1] = outa[0] + (size_t)BB * TDEC * TENC;
  outa[2] = outa[1] + (size_t)BB * TDEC * TENC;

  // ---- one-time preprocessing ----
  {
    size_t tot = (size_t)GATEN * KA;
    pack2_kernel<<<(unsigned)((tot + 255) / 256), 256, 0, stream>>>(
        W_ih_a, 512, W_hh_a, HC, WcatA, GATEN);
    tot = (size_t)GATEN * KD;
    pack2_kernel<<<(unsigned)((tot + 255) / 256), 256, 0, stream>>>(
        W_ih_d, 1536, W_hh_d, HC, WcatD, GATEN);
  }
  {
    int tot = TDEC * BB * PPC;
    prenet1_kernel<<<(tot + 255) / 256, 256, 0, stream>>>(dec_inputs, pw1, h1);
    prenet2_kernel<<<(tot + 255) / 256, 256, 0, stream>>>(h1, pw2, pre);
  }
  {
    int tot = BB * TENC * AAC;
    pmem_kernel<<<(tot + 255) / 256, 256, 0, stream>>>(mem1, Wm[0], pm[0], E1C);
    pmem_kernel<<<(tot + 255) / 256, 256, 0, stream>>>(mem2, Wm[1], pm[1], E2C);
    pmem_kernel<<<(tot + 255) / 256, 256, 0, stream>>>(mem3, Wm[2], pm[2], E3C);
  }
  zero_kernel<<<(zcount + 255) / 256, 256, 0, stream>>>(ha, zcount);

  // attention params
  Attn3 at;
  const float* mems[3] = {mem1, mem2, mem3};
  float* xs[3] = {x1, x2, x3};
  const int Es[3] = {E1C, E2C, E3C};
  for (int i = 0; i < 3; ++i) {
    at.p[i].Wq = Wq[i]; at.p[i].vv = vv[i]; at.p[i].lconv = lc[i];
    at.p[i].ldense = ld[i]; at.p[i].mem = mems[i]; at.p[i].pmem = pm[i];
    at.p[i].wprev = wprev[i]; at.p[i].wcum = wcum[i];
    at.p[i].x = xs[i]; at.p[i].aout = outa[i]; at.p[i].E = Es[i]; at.p[i].pad = 0;
  }

  Segs segA;  // [pre_t, x1, x2, x3] + ha   -> K = 1536
  segA.p[0] = nullptr; segA.len[0] = PPC;
  segA.p[1] = x1;      segA.len[1] = E1C;
  segA.p[2] = x2;      segA.len[2] = E2C;
  segA.p[3] = x3;      segA.len[3] = E3C;
  segA.p[4] = ha;      segA.len[4] = HC;
  segA.p[5] = nullptr; segA.len[5] = 0;

  Segs segD;  // [ha, x1, x2, x3, spk] + hd -> K = 2560
  segD.p[0] = ha;  segD.len[0] = HC;
  segD.p[1] = x1;  segD.len[1] = E1C;
  segD.p[2] = x2;  segD.len[2] = E2C;
  segD.p[3] = x3;  segD.len[3] = E3C;
  segD.p[4] = spk; segD.len[4] = SPKC;
  segD.p[5] = hd;  segD.len[5] = HC;

  // ---- 128 recurrent steps ----
  for (int t = 0; t < TDEC; ++t) {
    segA.p[0] = pre + (size_t)t * BB * PPC;
    wmma_gates_kernel<1536><<<32, 256, 0, stream>>>(WcatA, segA, b_ih_a, b_hh_a, gates);
    lstm_update_kernel<<<(BB * HC + 255) / 256, 256, 0, stream>>>(gates, ha, ca);
    attention_kernel<<<24, 256, 0, stream>>>(at, ha, t);
    wmma_gates_kernel<2560><<<32, 256, 0, stream>>>(WcatD, segD, b_ih_d, b_hh_d, gates);
    lstm_update_kernel<<<(BB * HC + 255) / 256, 256, 0, stream>>>(gates, hd, cd);
    proj_kernel<<<BB, 96, 0, stream>>>(hd, x1, x2, x3, spk, Wp, bp, Wg, bg,
                                       outmel, outgate, t);
  }
}
